// SurveyNeuralPredictor_91285234909623
// MI455X (gfx1250) — matrix operand order repaired and checked by reading, not compiled, not run on hardware
//
#include <hip/hip_runtime.h>
#include <math.h>

typedef __bf16 bf16;
typedef __attribute__((ext_vector_type(16))) __bf16 v16bf;
typedef __attribute__((ext_vector_type(8)))  __bf16 v8bf;
typedef __attribute__((ext_vector_type(8)))  float  v8f;

#define NE   1000000   // edges
#define NV   200000    // variables
#define DDIM 128       // variable_state dim
#define FEAT 16        // edge_feature dim
#define DIN  145       // 128 + 16 + 1
#define HID  256       // hidden dims (Hm = Hag = Ha = 256)

// swizzled bf16 weight sizes (elements): (ntiles, ksteps, 32 lanes, 16 elems)
#define W1MQ_N (16*5*32*16)   // [160(Kpad) x 256]
#define W2MQ_N (16*8*32*16)   // [256 x 256]
#define W1AQ_N (16*8*32*16)   // [256 x 256]
#define W2AQ_N (8*8*32*16)    // [256 x 128]

#define AGG_BYTES ((size_t)NV * HID * 4)

// ---------------- device helpers ----------------

__device__ __forceinline__ float logsig(float x) {
  // log-sigmoid(x) = min(x,0) - log(1 + exp(-|x|)).
  // Argument of log is in (1,2]: never denormal/negative, so use the raw
  // v_exp_f32 / v_log_f32 (base-2) ops and skip libm's denorm fixup chains.
  float t  = __builtin_amdgcn_exp2f(fabsf(x) * -1.44269504088896f); // exp(-|x|)
  float l2 = __builtin_amdgcn_logf(1.0f + t);                       // log2(1+t)
  return fminf(x, 0.0f) - 0.693147180559945f * l2;
}

__device__ __forceinline__ v8f zero8() {
  v8f z;
#pragma unroll
  for (int i = 0; i < 8; i++) z[i] = 0.0f;
  return z;
}

__device__ __forceinline__ v8f wmma_bf16(v16bf a, v16bf b, v8f c) {
  // (neg_a, A, neg_b, B, c_mod, C, reuse_a, reuse_b)
  return __builtin_amdgcn_wmma_f32_16x16x32_bf16(false, a, false, b, (short)0, c,
                                                 false, false);
}

// CDNA5 async copy: global -> LDS, 16B per lane, tracked by ASYNCcnt.
// GV mode: first operand = LDS byte-offset VGPR, second = 64-bit address pair.
__device__ __forceinline__ void async_g2l_b128(unsigned lds_off, const void* gaddr) {
  asm volatile("global_load_async_to_lds_b128 %0, %1, off"
               :: "v"(lds_off), "v"(gaddr) : "memory");
}
__device__ __forceinline__ void wait_asynccnt0() {
  asm volatile("s_wait_asynccnt 0x0" ::: "memory");
}

// ---------------- kernel: zero agg ----------------

__global__ void k_zero(float4* __restrict__ p, int n4) {
  int i = blockIdx.x * blockDim.x + threadIdx.x;
  const int stride = gridDim.x * blockDim.x;
  float4 z; z.x = 0.f; z.y = 0.f; z.z = 0.f; z.w = 0.f;
  for (; i < n4; i += stride) p[i] = z;
}

// ---------------- kernel: convert weights to bf16, WMMA-B-fragment swizzle ---
// B-fragment layout (16-bit B, 32x16): lane l -> n = l&15, h = l>>4;
// element j -> K = kstep*32 + (j&7) + 8*h + 16*(j>>3). Fragment data stored contiguously:
// idx = ((ntile*KS + ks)*32 + lane)*16 + j

__global__ void k_conv(const float* __restrict__ W1m, const float* __restrict__ W2m,
                       const float* __restrict__ W1a, const float* __restrict__ W2a,
                       bf16* __restrict__ w1mq, bf16* __restrict__ w2mq,
                       bf16* __restrict__ w1aq, bf16* __restrict__ w2aq) {
  int idx = blockIdx.x * blockDim.x + threadIdx.x;
  if (idx < W1MQ_N) {
    int j = idx & 15, lane = (idx >> 4) & 31, f = idx >> 9;
    int ks = f % 5, nt = f / 5;
    int n = nt * 16 + (lane & 15);
    int k = ks * 32 + (j & 7) + 8 * (lane >> 4) + 16 * (j >> 3);
    w1mq[idx] = (k < DIN) ? (bf16)W1m[n * DIN + k] : (bf16)0.0f;
    return;
  }
  idx -= W1MQ_N;
  if (idx < W2MQ_N) {
    int j = idx & 15, lane = (idx >> 4) & 31, f = idx >> 9;
    int ks = f & 7, nt = f >> 3;
    int n = nt * 16 + (lane & 15);
    int k = ks * 32 + (j & 7) + 8 * (lane >> 4) + 16 * (j >> 3);
    w2mq[idx] = (bf16)W2m[n * HID + k];
    return;
  }
  idx -= W2MQ_N;
  if (idx < W1AQ_N) {
    int j = idx & 15, lane = (idx >> 4) & 31, f = idx >> 9;
    int ks = f & 7, nt = f >> 3;
    int n = nt * 16 + (lane & 15);
    int k = ks * 32 + (j & 7) + 8 * (lane >> 4) + 16 * (j >> 3);
    w1aq[idx] = (bf16)W1a[n * HID + k];
    return;
  }
  idx -= W1AQ_N;
  if (idx < W2AQ_N) {
    int j = idx & 15, lane = (idx >> 4) & 31, f = idx >> 9;
    int ks = f & 7, nt = f >> 3;     // nt in 0..7 (D = 128 outputs)
    int n = nt * 16 + (lane & 15);
    int k = ks * 32 + (j & 7) + 8 * (lane >> 4) + 16 * (j >> 3);
    w2aq[idx] = (bf16)W2a[n * HID + k];
  }
}

// ---------------- kernel 1: edge MLP + atomic segment-sum ----------------
// LDS layout (dynamic, 280064 B < 320 KB WGP LDS):

#define K1_W1_OFF 0
#define K1_W2_OFF (K1_W1_OFF + W1MQ_N*2)          // 81920
#define K1_B1_OFF (K1_W2_OFF + W2MQ_N*2)          // 212992
#define K1_S_OFF  (K1_B1_OFF + 1024)              // 214016
#define K1_VX_OFF (K1_S_OFF + 8*16*HID*2)         // 279552
#define K1_SMEM   (K1_VX_OFF + 8*16*4)            // 280064

__global__ __launch_bounds__(256, 1)
void k_edge(const float* __restrict__ vs, const float* __restrict__ ef,
            const float* __restrict__ sol, const float* __restrict__ b1m,
            const int* __restrict__ vidx, const int* __restrict__ esign,
            const bf16* __restrict__ w1q, const bf16* __restrict__ w2q,
            float* __restrict__ agg) {
  extern __shared__ char smem[];
  bf16*  w1    = (bf16*)(smem + K1_W1_OFF);
  bf16*  w2    = (bf16*)(smem + K1_W2_OFF);
  float* bias1 = (float*)(smem + K1_B1_OFF);
  bf16*  sbuf  = (bf16*)(smem + K1_S_OFF);
  int*   vx    = (int*)(smem + K1_VX_OFF);

  const int tid = threadIdx.x;
  {  // async cooperative weight load into LDS (once per persistent block)
    for (int i = tid; i < (W1MQ_N * 2) / 16; i += 256)
      async_g2l_b128(K1_W1_OFF + 16u * i, (const char*)w1q + (size_t)i * 16);
    for (int i = tid; i < (W2MQ_N * 2) / 16; i += 256)
      async_g2l_b128(K1_W2_OFF + 16u * i, (const char*)w2q + (size_t)i * 16);
    if (tid < HID) bias1[tid] = b1m[tid];
    wait_asynccnt0();
  }
  __syncthreads();

  const int w    = tid >> 5;      // wave id (8 waves)
  const int lane = tid & 31;
  const int h    = lane >> 4;
  const int nl   = lane & 15;
  const int m    = lane & 15;     // row within 16-edge tile
  bf16* sw  = sbuf + w * (16 * HID);
  int*  vxw = vx + w * 16;

  const int numTiles = NE / 16;   // 62500, exact
  for (int tile = blockIdx.x * 8 + w; tile < numTiles; tile += (int)gridDim.x * 8) {
    const int ebase = tile * 16;
    if (lane < 16) vxw[lane] = vidx[ebase + lane];
    {
      int tnext = tile + (int)gridDim.x * 8;
      if (tnext < numTiles && lane < 16)
        __builtin_prefetch(vs + (size_t)(tnext * 16 + lane) * DDIM, 0, 1);
    }
    const int e = ebase + m;

    // ---- A fragments of x = [variable_state | edge_feature | signed | pad0] ----
    // A 16-bit 16x32 layout: elem j -> K = ks*32 + 8h + j  (j<8)
    //                                 K = ks*32 + 8h + 8 + j (j>=8), i.e. +16 offset
    v16bf afrag[5];
#pragma unroll
    for (int ks = 0; ks < 5; ks++) {
      v16bf a;
#pragma unroll
      for (int half = 0; half < 2; half++) {
        const int S = ks * 32 + 8 * h + 16 * half;   // start of 8-element run
        float t[8];
        if (S + 8 <= DDIM) {
          const float4* p = (const float4*)(vs + (size_t)e * DDIM + S);
          float4 x0 = p[0], x1 = p[1];
          t[0] = x0.x; t[1] = x0.y; t[2] = x0.z; t[3] = x0.w;
          t[4] = x1.x; t[5] = x1.y; t[6] = x1.z; t[7] = x1.w;
        } else if (S < DDIM + FEAT) {                // S == 128 or 136
          const float4* p = (const float4*)(ef + (size_t)e * FEAT + (S - DDIM));
          float4 x0 = p[0], x1 = p[1];
          t[0] = x0.x; t[1] = x0.y; t[2] = x0.z; t[3] = x0.w;
          t[4] = x1.x; t[5] = x1.y; t[6] = x1.z; t[7] = x1.w;
        } else if (S == DDIM + FEAT) {               // S == 144: signed solution col
          float sg = (float)esign[e];
          t[0] = sg * sol[vxw[m]];
#pragma unroll
          for (int q = 1; q < 8; q++) t[q] = 0.0f;
        } else {                                     // zero pad (K >= 145)
#pragma unroll
          for (int q = 0; q < 8; q++) t[q] = 0.0f;
        }
#pragma unroll
        for (int j = 0; j < 8; j++) a[half * 8 + j] = (bf16)t[j];
      }
      afrag[ks] = a;
    }

    // ---- GEMM1: [16x160] x [160x256], +bias, logsigmoid -> s (bf16, LDS) ----
    // Two N-tiles in flight: two independent WMMA accumulator chains (2x ILP
    // in the matrix pipe; back-to-back dependent WMMAs otherwise stall).
    for (int nt = 0; nt < 16; nt += 2) {
      v8f acc0 = zero8(), acc1 = zero8();
#pragma unroll
      for (int ks = 0; ks < 5; ks++) {
        v16bf b0 = *(const v16bf*)(w1 + ((size_t)(((nt + 0) * 5 + ks) * 32 + lane)) * 16);
        v16bf b1 = *(const v16bf*)(w1 + ((size_t)(((nt + 1) * 5 + ks) * 32 + lane)) * 16);
        acc0 = wmma_bf16(afrag[ks], b0, acc0);
        acc1 = wmma_bf16(afrag[ks], b1, acc1);
      }
      const float bn0 = bias1[(nt + 0) * 16 + nl];
      const float bn1 = bias1[(nt + 1) * 16 + nl];
#pragma unroll
      for (int r = 0; r < 8; r++) {
        float v0 = logsig(acc0[r] + bn0);
        float v1 = logsig(acc1[r] + bn1);
        sw[(r + 8 * h) * HID + (nt + 0) * 16 + nl] = (bf16)v0; // C elem (M=r+8h,N=nl)
        sw[(r + 8 * h) * HID + (nt + 1) * 16 + nl] = (bf16)v1;
      }
    }

    // ---- reload s as A fragments ----
    v16bf a2[8];
#pragma unroll
    for (int ks = 0; ks < 8; ks++) {
      const int S = ks * 32 + 8 * h;
      v8bf lo = *(const v8bf*)(sw + m * HID + S);
      v8bf hi = *(const v8bf*)(sw + m * HID + S + 16);
      v16bf a;
#pragma unroll
      for (int j = 0; j < 8; j++) { a[j] = lo[j]; a[j + 8] = hi[j]; }
      a2[ks] = a;
    }

    // ---- GEMM2: [16x256] x [256x256], logsigmoid, atomic segment-sum ----
    for (int nt = 0; nt < 16; nt += 2) {
      v8f acc0 = zero8(), acc1 = zero8();
#pragma unroll
      for (int ks = 0; ks < 8; ks++) {
        v16bf b0 = *(const v16bf*)(w2 + ((size_t)(((nt + 0) * 8 + ks) * 32 + lane)) * 16);
        v16bf b1 = *(const v16bf*)(w2 + ((size_t)(((nt + 1) * 8 + ks) * 32 + lane)) * 16);
        acc0 = wmma_bf16(a2[ks], b0, acc0);
        acc1 = wmma_bf16(a2[ks], b1, acc1);
      }
#pragma unroll
      for (int r = 0; r < 8; r++) {
        float v0 = logsig(acc0[r]);
        float v1 = logsig(acc1[r]);
        int row = vxw[r + 8 * h];
        unsafeAtomicAdd(agg + (size_t)row * HID + (nt + 0) * 16 + nl, v0);
        unsafeAtomicAdd(agg + (size_t)row * HID + (nt + 1) * 16 + nl, v1);
      }
    }
  }
}

// ---------------- kernel 2: aggregator MLP + classifier ----------------

#define K2_W1_OFF 0
#define K2_W2_OFF (K2_W1_OFF + W1AQ_N*2)   // 131072
#define K2_B1_OFF (K2_W2_OFF + W2AQ_N*2)   // 196608
#define K2_WC_OFF (K2_B1_OFF + 1024)       // 197632
#define K2_S_OFF  (K2_WC_OFF + 1024)       // 198656
#define K2_SMEM   (K2_S_OFF + 8*16*HID*2)  // 264192

__global__ __launch_bounds__(256, 1)
void k_var(const float* __restrict__ agg, const float* __restrict__ b1a,
           const float* __restrict__ Wc, const float* __restrict__ bc,
           const bf16* __restrict__ w1q, const bf16* __restrict__ w2q,
           float* __restrict__ out) {
  extern __shared__ char smem[];
  bf16*  w1    = (bf16*)(smem + K2_W1_OFF);
  bf16*  w2    = (bf16*)(smem + K2_W2_OFF);
  float* bias1 = (float*)(smem + K2_B1_OFF);
  float* wc    = (float*)(smem + K2_WC_OFF);
  bf16*  sbuf  = (bf16*)(smem + K2_S_OFF);

  const int tid = threadIdx.x;
  {
    for (int i = tid; i < (W1AQ_N * 2) / 16; i += 256)
      async_g2l_b128(K2_W1_OFF + 16u * i, (const char*)w1q + (size_t)i * 16);
    for (int i = tid; i < (W2AQ_N * 2) / 16; i += 256)
      async_g2l_b128(K2_W2_OFF + 16u * i, (const char*)w2q + (size_t)i * 16);
    if (tid < HID) bias1[tid] = b1a[tid];
    wc[tid] = Wc[tid];   // 256 floats: [p=0: d<128][p=1: d<128]
    wait_asynccnt0();
  }
  __syncthreads();

  const int w = tid >> 5, lane = tid & 31, h = lane >> 4, nl = lane & 15, m = lane & 15;
  bf16* sw = sbuf + w * (16 * HID);
  const float bc0 = bc[0], bc1 = bc[1];

  const int numTiles = NV / 16;   // 12500, exact
  for (int tile = blockIdx.x * 8 + w; tile < numTiles; tile += (int)gridDim.x * 8) {
    const int vbase = tile * 16;
    const size_t rowOff = (size_t)(vbase + m) * HID;

    // A fragments from agg rows (f32 -> bf16)
    v16bf a1[8];
#pragma unroll
    for (int ks = 0; ks < 8; ks++) {
      v16bf a;
#pragma unroll
      for (int half = 0; half < 2; half++) {
        const int S = ks * 32 + 8 * h + 16 * half;
        const float4* p = (const float4*)(agg + rowOff + S);
        float4 x0 = p[0], x1 = p[1];
        a[half * 8 + 0] = (bf16)x0.x; a[half * 8 + 1] = (bf16)x0.y;
        a[half * 8 + 2] = (bf16)x0.z; a[half * 8 + 3] = (bf16)x0.w;
        a[half * 8 + 4] = (bf16)x1.x; a[half * 8 + 5] = (bf16)x1.y;
        a[half * 8 + 6] = (bf16)x1.z; a[half * 8 + 7] = (bf16)x1.w;
      }
      a1[ks] = a;
    }

    // GEMM1: [16x256] x [256x256], +bias, logsigmoid -> s (dual acc chains)
    for (int nt = 0; nt < 16; nt += 2) {
      v8f acc0 = zero8(), acc1 = zero8();
#pragma unroll
      for (int ks = 0; ks < 8; ks++) {
        v16bf b0 = *(const v16bf*)(w1 + ((size_t)(((nt + 0) * 8 + ks) * 32 + lane)) * 16);
        v16bf b1 = *(const v16bf*)(w1 + ((size_t)(((nt + 1) * 8 + ks) * 32 + lane)) * 16);
        acc0 = wmma_bf16(a1[ks], b0, acc0);
        acc1 = wmma_bf16(a1[ks], b1, acc1);
      }
      const float bn0 = bias1[(nt + 0) * 16 + nl];
      const float bn1 = bias1[(nt + 1) * 16 + nl];
#pragma unroll
      for (int r = 0; r < 8; r++) {
        sw[(r + 8 * h) * HID + (nt + 0) * 16 + nl] = (bf16)logsig(acc0[r] + bn0);
        sw[(r + 8 * h) * HID + (nt + 1) * 16 + nl] = (bf16)logsig(acc1[r] + bn1);
      }
    }

    // reload as A fragments
    v16bf a2[8];
#pragma unroll
    for (int ks = 0; ks < 8; ks++) {
      const int S = ks * 32 + 8 * h;
      v8bf lo = *(const v8bf*)(sw + m * HID + S);
      v8bf hi = *(const v8bf*)(sw + m * HID + S + 16);
      v16bf a;
#pragma unroll
      for (int j = 0; j < 8; j++) { a[j] = lo[j]; a[j + 8] = hi[j]; }
      a2[ks] = a;
    }

    // GEMM2: [16x256] x [256x128], logsigmoid, fused classifier partials
    float c0[8], c1[8];
#pragma unroll
    for (int r = 0; r < 8; r++) { c0[r] = 0.f; c1[r] = 0.f; }
    for (int nt = 0; nt < 8; nt += 2) {
      v8f acc0 = zero8(), acc1 = zero8();
#pragma unroll
      for (int ks = 0; ks < 8; ks++) {
        v16bf b0 = *(const v16bf*)(w2 + ((size_t)(((nt + 0) * 8 + ks) * 32 + lane)) * 16);
        v16bf b1 = *(const v16bf*)(w2 + ((size_t)(((nt + 1) * 8 + ks) * 32 + lane)) * 16);
        acc0 = wmma_bf16(a2[ks], b0, acc0);
        acc1 = wmma_bf16(a2[ks], b1, acc1);
      }
      const int d0 = (nt + 0) * 16 + nl;
      const int d1 = (nt + 1) * 16 + nl;
      const float wc00 = wc[d0], wc10 = wc[128 + d0];
      const float wc01 = wc[d1], wc11 = wc[128 + d1];
#pragma unroll
      for (int r = 0; r < 8; r++) {
        float h0 = logsig(acc0[r]);
        float h1 = logsig(acc1[r]);
        c0[r] += h0 * wc00 + h1 * wc01;
        c1[r] += h0 * wc10 + h1 * wc11;
      }
    }
    // cross-lane reduce over the 16 lanes sharing h, then write [V,2]
#pragma unroll
    for (int r = 0; r < 8; r++) {
#pragma unroll
      for (int off = 8; off > 0; off >>= 1) {
        c0[r] += __shfl_xor(c0[r], off, 32);
        c1[r] += __shfl_xor(c1[r], off, 32);
      }
      if (nl == 0) {
        const int row = vbase + r + 8 * h;
        out[row * 2 + 0] = c0[r] + bc0;
        out[row * 2 + 1] = c1[r] + bc1;
      }
    }
  }
}

// ---------------- host launcher ----------------

extern "C" void kernel_launch(void* const* d_in, const int* in_sizes, int n_in,
                              void* d_out, int out_size, void* d_ws, size_t ws_size,
                              hipStream_t stream) {
  (void)in_sizes; (void)n_in; (void)out_size; (void)ws_size;
  const float* vs   = (const float*)d_in[0];   // variable_state [E,128]
  const float* ef   = (const float*)d_in[1];   // edge_feature  [E,16]
  const float* sol  = (const float*)d_in[2];   // solution [V]
  const float* W1m  = (const float*)d_in[3];   // [256,145]
  const float* b1m  = (const float*)d_in[4];   // [256]
  const float* W2m  = (const float*)d_in[5];   // [256,256]
  const float* W1a  = (const float*)d_in[6];   // [256,256]
  const float* b1a  = (const float*)d_in[7];   // [256]
  const float* W2a  = (const float*)d_in[8];   // [128,256]
  const float* Wc   = (const float*)d_in[9];   // [2,128]
  const float* bc   = (const float*)d_in[10];  // [2]
  const int* vidx   = (const int*)d_in[11];    // [E]
  const int* esign  = (const int*)d_in[12];    // [E]
  float* out = (float*)d_out;

  char* ws = (char*)d_ws;
  float* agg  = (float*)(ws);
  bf16* w1mq = (bf16*)(ws + AGG_BYTES);
  bf16* w2mq = (bf16*)((char*)w1mq + (size_t)W1MQ_N * 2);
  bf16* w1aq = (bf16*)((char*)w2mq + (size_t)W2MQ_N * 2);
  bf16* w2aq = (bf16*)((char*)w1aq + (size_t)W1AQ_N * 2);

  // allow >64KB dynamic LDS (CDNA5 WGP: 320 KB). Idempotent; capture-safe.
  hipFuncSetAttribute((const void*)k_edge,
                      hipFuncAttributeMaxDynamicSharedMemorySize, K1_SMEM);
  hipFuncSetAttribute((const void*)k_var,
                      hipFuncAttributeMaxDynamicSharedMemorySize, K2_SMEM);

  k_zero<<<2048, 256, 0, stream>>>((float4*)agg, (NV * HID) / 4);
  {
    int total = W1MQ_N + W2MQ_N + W1AQ_N + W2AQ_N;
    k_conv<<<(total + 255) / 256, 256, 0, stream>>>(W1m, W2m, W1a, W2a,
                                                    w1mq, w2mq, w1aq, w2aq);
  }
  k_edge<<<1024, 256, K1_SMEM, stream>>>(vs, ef, sol, b1m, vidx, esign,
                                         w1mq, w2mq, agg);
  k_var<<<512, 256, K2_SMEM, stream>>>(agg, b1a, Wc, bc, w1aq, w2aq, out);
}
